// Transformer3D_61220463837494
// MI455X (gfx1250) — matrix operand order repaired and checked by reading, not compiled, not run on hardware
//
#include <hip/hip_runtime.h>
#include <hip/hip_bf16.h>

// ---------------- problem constants (from reference) ----------------
#define C_DIM   256
#define NHEADS  8
#define DKH     32      // dk = C/h
#define SEQ_N   4096    // keys / sources
#define SEQ_N2  1024    // queries / targets
#define BATCH   4
#define NEARK   5
#define KT_TILE 64      // keys per flash-attention tile (4 S-fragments)
#define QWAVES  4       // waves (q-tiles) per attention workgroup

typedef __attribute__((ext_vector_type(16))) __bf16       v16bf;
typedef __attribute__((ext_vector_type(8)))  float        v8f;
typedef __attribute__((ext_vector_type(8)))  unsigned int v8u;   // 32B fragment loads

struct TwoU4 { uint4 a, b; };

// contiguous 32B -> bf16 fragment (requires 32B alignment)
__device__ __forceinline__ v16bf frag_ld32(const __bf16* p) {
    return __builtin_bit_cast(v16bf, *reinterpret_cast<const v8u*>(p));
}
// two 16B chunks -> bf16 fragment
__device__ __forceinline__ v16bf frag_ld2x16(const __bf16* p0, const __bf16* p1) {
    TwoU4 u;
    u.a = *reinterpret_cast<const uint4*>(p0);
    u.b = *reinterpret_cast<const uint4*>(p1);
    return __builtin_bit_cast(v16bf, u);
}

// ---------------- async global->LDS copy (16B), guarded ----------------
#if defined(__has_builtin)
#  if __has_builtin(__builtin_amdgcn_global_load_async_to_lds_b128)
#    define HAVE_ASYNC_LDS 1
#  endif
#endif
#ifndef HAVE_ASYNC_LDS
#  define HAVE_ASYNC_LDS 0
#endif

__device__ __forceinline__ void async_copy16(const void* g, void* l) {
#if HAVE_ASYNC_LDS
    // param0: v4i in global AS (64-bit flat value == global address)
    // param1: v4i in LDS AS    (low 32 bits of generic shared ptr == LDS offset)
    typedef __attribute__((__vector_size__(4 * sizeof(int)))) int v4i_t;
    typedef v4i_t __attribute__((address_space(1)))* gv4_p;
    typedef v4i_t __attribute__((address_space(3)))* lv4_p;
    typedef unsigned long long uptr_t;
    __builtin_amdgcn_global_load_async_to_lds_b128(
        (gv4_p)(uptr_t)g, (lv4_p)(unsigned)(uptr_t)l, 0, 0);
#else
    *reinterpret_cast<uint4*>(l) = *reinterpret_cast<const uint4*>(g);
#endif
}
__device__ __forceinline__ void wait_async0() {
#if defined(__has_builtin) && __has_builtin(__builtin_amdgcn_s_wait_asynccnt)
    __builtin_amdgcn_s_wait_asynccnt(0);
#else
    asm volatile("s_wait_asynccnt 0" ::: "memory");
#endif
}

// =====================================================================
// 1) QKV projection: Y_bf16 = X[S,B,C] @ W[C,C] + bias
//    TR==0 : [B,h,S,dk] (Q,K)    TR==1 : [B,h,dk,S] (V transposed)
// =====================================================================
template <int TR>
__global__ __launch_bounds__(32) void proj_kernel(
    const float* __restrict__ X, const float* __restrict__ W,
    const float* __restrict__ bias, __bf16* __restrict__ Y, int S)
{
    const int lane = threadIdx.x & 31;
    const int l15  = lane & 15;
    const int half = lane >> 4;
    const int s0   = blockIdx.x * 16;
    const int c0   = blockIdx.y * 16;
    const int b    = blockIdx.z;

    const int arow = s0 + l15;
    const int kbA  = half * 16;       // A: lanes<16 K 0-15; lanes>=16 K 16-31
    const int colB = c0 + l15;
    const int kbB  = half * 16;       // B: lanes<16 K 0-15; lanes>=16 K 16-31

    v8f acc = {};
    for (int k0 = 0; k0 < C_DIM; k0 += 32) {
        v16bf a, bm;
        const float* xp = X + ((long)arow * BATCH + b) * C_DIM + k0;
#pragma unroll
        for (int j = 0; j < 8; ++j) {
            a[j]     = (__bf16)xp[kbA + j];
            a[8 + j] = (__bf16)xp[kbA + 8 + j];
        }
#pragma unroll
        for (int j = 0; j < 16; ++j)
            bm[j] = (__bf16)W[(long)(k0 + kbB + j) * C_DIM + colB];
        acc = __builtin_amdgcn_wmma_f32_16x16x32_bf16(false, a, false, bm,
                                                      (short)0, acc, false, false);
    }

    const float bb = bias[colB];
    const int hh = colB >> 5, dd = colB & 31;
#pragma unroll
    for (int r = 0; r < 8; ++r) {
        int srow = s0 + r + half * 8;
        float v = acc[r] + bb;
        long idx = TR ? ((long)((b * NHEADS + hh) * DKH + dd) * S + srow)
                      : ((long)((b * NHEADS + hh) * S + srow) * DKH + dd);
        Y[idx] = (__bf16)v;
    }
}

// =====================================================================
// 2) per-source top-5 nearest targets -> packed 64B records per key:
//    dw[0..3]=idx0-3, dw[4..7]=w0-3 (f32 bits), dw[8]=idx4, dw[9]=w4
// =====================================================================
__global__ void topk_kernel(const float* __restrict__ sp,
                            const float* __restrict__ tp,
                            int* __restrict__ nnp)
{
    int t = blockIdx.x * blockDim.x + threadIdx.x;
    if (t >= BATCH * SEQ_N) return;
    int b = t / SEQ_N, i = t % SEQ_N;

    float sx = sp[((long)i * BATCH + b) * 3 + 0];
    float sy = sp[((long)i * BATCH + b) * 3 + 1];
    float sz = sp[((long)i * BATCH + b) * 3 + 2];

    float bd[NEARK]; int bi[NEARK];
#pragma unroll
    for (int k = 0; k < NEARK; ++k) { bd[k] = 3.0e38f; bi[k] = 0; }

    for (int j = 0; j < SEQ_N2; ++j) {
        const float* q = tp + ((long)j * BATCH + b) * 3;
        float dx = sx - q[0], dy = sy - q[1], dz = sz - q[2];
        float d = dx * dx + dy * dy + dz * dz;
        if (d < bd[NEARK - 1]) {
            int p = NEARK - 1;
            while (p > 0 && bd[p - 1] > d) { bd[p] = bd[p - 1]; bi[p] = bi[p - 1]; --p; }
            bd[p] = d; bi[p] = j;
        }
    }
    float r[NEARK], s = 0.f;
#pragma unroll
    for (int k = 0; k < NEARK; ++k) { r[k] = 1.0f / (sqrtf(bd[k]) + 0.1f); s += r[k]; }

    int* rec = nnp + ((long)b * SEQ_N + i) * 16;
#pragma unroll
    for (int k = 0; k < 4; ++k) {
        rec[k]     = bi[k];
        rec[4 + k] = __builtin_bit_cast(int, r[k] / s);
    }
    rec[8] = bi[4];
    rec[9] = __builtin_bit_cast(int, r[4] / s);
}

// =====================================================================
// 3) flash attention: 4-wave workgroup, one q-tile (16 queries) per wave,
//    shared (b,h). 64-key tiles staged into double-buffered LDS with
//    async global->LDS copies (ASYNCcnt) overlapped with the 8 WMMAs of
//    the previous tile; workgroup barriers publish the staged tiles.
// =====================================================================
__global__ __launch_bounds__(32 * QWAVES) void attn_kernel(
    const __bf16* __restrict__ Qb, const __bf16* __restrict__ Kb,
    const __bf16* __restrict__ Vt, const int* __restrict__ nnp,
    float* __restrict__ O)
{
    __shared__ __align__(32) __bf16 KT[2][KT_TILE * DKH];      // [buf][key*32+d]  8KB
    __shared__ __align__(32) __bf16 VT[2][DKH * KT_TILE];      // [buf][d*64+k]    8KB
    __shared__ __align__(32) __bf16 Pst[QWAVES][16 * 72];      // per-wave P tile  9KB

    const int tid  = threadIdx.x;
    const int wid  = tid >> 5;
    const int lane = tid & 31;
    const int l15  = lane & 15;
    const int half = lane >> 4;
    const int q0   = (blockIdx.x * QWAVES + wid) * 16;
    const int hh   = blockIdx.y;
    const int b    = blockIdx.z;
    const int kb   = half * 16;       // A-frag K offsets
    const int kb2  = half * 16;       // B-frag K offsets
    const float scale = 0.17677669529663687f;    // 1/sqrt(dk=32)

    const __bf16* kbase = Kb + (long)(b * NHEADS + hh) * SEQ_N * DKH;
    const __bf16* vbase = Vt + (long)(b * NHEADS + hh) * DKH * SEQ_N;

    // cooperative staging of one 64-key K tile (4KB) + V tile (4KB)
    auto stage = [&](int buf, int kt) {
        const char* gK = (const char*)(kbase + (long)kt * DKH);   // contiguous 4KB
        char*       lK = (char*)&KT[buf][0];
        async_copy16(gK + tid * 16,           lK + tid * 16);
        async_copy16(gK + (tid + 128) * 16,   lK + (tid + 128) * 16);
#pragma unroll
        for (int c = tid; c < 256; c += 128) {                    // V: 32 rows x 128B
            int d = c >> 3, o = (c & 7) * 16;
            async_copy16((const char*)(vbase + (long)d * SEQ_N + kt) + o,
                         (char*)&VT[buf][d * KT_TILE] + o);
        }
    };

    // Q fragment (A layout), loaded once per wave
    const __bf16* qp = Qb + ((long)(b * NHEADS + hh) * SEQ_N2 + (q0 + l15)) * DKH;
    v16bf qa = frag_ld2x16(qp + kb, qp + kb + 8);

    float m[8], l[8];
    v8f o0 = {}, o1 = {};
#pragma unroll
    for (int e = 0; e < 8; ++e) { m[e] = -3.0e38f; l[e] = 0.0f; }

    const int rbase = q0 + half * 8;
    const v8f zero  = {};
    __bf16* pst = &Pst[wid][0];

    stage(0, 0);                                   // fill first buffer

    int buf = 0;
    for (int kt = 0; kt < SEQ_N; kt += KT_TILE, buf ^= 1) {
        wait_async0();                             // my staged copies landed
        asm volatile("s_wait_dscnt 0" ::: "memory");
        __syncthreads();                           // everyone's copies visible

        if (kt + KT_TILE < SEQ_N) stage(buf ^ 1, kt + KT_TILE);   // overlap next tile

        // ---- S = Q K^T : 4 fragments of 16 keys each, from LDS ----
        v8f s[4];
#pragma unroll
        for (int f = 0; f < 4; ++f)
            s[f] = __builtin_amdgcn_wmma_f32_16x16x32_bf16(
                       false, qa, false,
                       frag_ld32(&KT[buf][(f * 16 + l15) * DKH + kb2]),
                       (short)0, zero, false, false);

        // ---- this lane's 4 key columns: packed top-5 (target, weight) ----
        int dlt[4][NEARK]; float wv[4][NEARK];
#pragma unroll
        for (int f = 0; f < 4; ++f) {
            const int* rec = nnp + ((long)b * SEQ_N + (kt + f * 16 + l15)) * 16;
            uint4 i03 = *(const uint4*)rec;
            uint4 w03 = *(const uint4*)(rec + 4);
            int2  t4  = *(const int2*)(rec + 8);
            dlt[f][0] = (int)i03.x - rbase;  wv[f][0] = __builtin_bit_cast(float, w03.x);
            dlt[f][1] = (int)i03.y - rbase;  wv[f][1] = __builtin_bit_cast(float, w03.y);
            dlt[f][2] = (int)i03.z - rbase;  wv[f][2] = __builtin_bit_cast(float, w03.z);
            dlt[f][3] = (int)i03.w - rbase;  wv[f][3] = __builtin_bit_cast(float, w03.w);
            dlt[f][4] = t4.x - rbase;        wv[f][4] = __builtin_bit_cast(float, t4.y);
        }

        // ---- online softmax over the 64-key tile ----
#pragma unroll
        for (int e = 0; e < 8; ++e) {
            float v[4];
#pragma unroll
            for (int f = 0; f < 4; ++f) {
                float mv = -1.0e9f;                // mask[b, q0+e+half*8, key]
#pragma unroll
                for (int t = 0; t < NEARK; ++t)
                    if (dlt[f][t] == e) mv = wv[f][t];
                v[f] = s[f][e] * scale + mv;
            }
            float tmax = fmaxf(fmaxf(v[0], v[1]), fmaxf(v[2], v[3]));
#pragma unroll
            for (int off = 1; off < 16; off <<= 1)
                tmax = fmaxf(tmax, __shfl_xor(tmax, off, 32));
            float newm = fmaxf(m[e], tmax);
            float p[4], rs = 0.f;
#pragma unroll
            for (int f = 0; f < 4; ++f) { p[f] = __expf(v[f] - newm); rs += p[f]; }
#pragma unroll
            for (int off = 1; off < 16; off <<= 1)
                rs += __shfl_xor(rs, off, 32);
            float alpha = __expf(m[e] - newm);
            l[e] = l[e] * alpha + rs;
            m[e] = newm;
            o0[e] *= alpha;
            o1[e] *= alpha;

            int prow = e + half * 8;               // stage P in C layout
#pragma unroll
            for (int f = 0; f < 4; ++f)
                pst[prow * 72 + f * 16 + l15] = (__bf16)p[f];
        }

        // order P stores before the A-layout reload (single wave, DScnt)
        asm volatile("s_wait_dscnt 0" ::: "memory");

        const __bf16* pp = pst + l15 * 72;
        v16bf pa0 = frag_ld2x16(pp + kb,      pp + 8 + kb);
        v16bf pa1 = frag_ld2x16(pp + 32 + kb, pp + 40 + kb);

        // ---- O += P V : V tile is [dk][64] in LDS, B-frags contiguous ----
        const __bf16* v00 = &VT[buf][l15 * KT_TILE + kb2];
        const __bf16* v10 = &VT[buf][(16 + l15) * KT_TILE + kb2];
        o0 = __builtin_amdgcn_wmma_f32_16x16x32_bf16(false, pa0, false, frag_ld32(v00),      (short)0, o0, false, false);
        o0 = __builtin_amdgcn_wmma_f32_16x16x32_bf16(false, pa1, false, frag_ld32(v00 + 32), (short)0, o0, false, false);
        o1 = __builtin_amdgcn_wmma_f32_16x16x32_bf16(false, pa0, false, frag_ld32(v10),      (short)0, o1, false, false);
        o1 = __builtin_amdgcn_wmma_f32_16x16x32_bf16(false, pa1, false, frag_ld32(v10 + 32), (short)0, o1, false, false);

        __syncthreads();                           // all reads done before re-stage
    }

    // normalize and write O as [B, N2, h*dk] fp32
#pragma unroll
    for (int e = 0; e < 8; ++e) {
        int row = q0 + e + half * 8;
        float inv = 1.0f / l[e];
        long ob = ((long)(b * SEQ_N2 + row) * NHEADS + hh) * DKH;
        O[ob + l15]      = o0[e] * inv;
        O[ob + 16 + l15] = o1[e] * inv;
    }
}

// =====================================================================
// 4) output projection: out[q,b,:] = O[b,q,:] @ Wo + bo   (fp32 out)
// =====================================================================
__global__ __launch_bounds__(32) void outproj_kernel(
    const float* __restrict__ Oin, const float* __restrict__ W,
    const float* __restrict__ bias, float* __restrict__ out)
{
    const int lane = threadIdx.x & 31;
    const int l15  = lane & 15;
    const int half = lane >> 4;
    const int s0   = blockIdx.x * 16;
    const int c0   = blockIdx.y * 16;
    const int b    = blockIdx.z;

    const int arow = s0 + l15;
    const int kbA  = half * 16;
    const int colB = c0 + l15;
    const int kbB  = half * 16;

    v8f acc = {};
    for (int k0 = 0; k0 < C_DIM; k0 += 32) {
        v16bf a, bm;
        const float* xp = Oin + ((long)b * SEQ_N2 + arow) * C_DIM + k0;
#pragma unroll
        for (int j = 0; j < 8; ++j) {
            a[j]     = (__bf16)xp[kbA + j];
            a[8 + j] = (__bf16)xp[kbA + 8 + j];
        }
#pragma unroll
        for (int j = 0; j < 16; ++j)
            bm[j] = (__bf16)W[(long)(k0 + kbB + j) * C_DIM + colB];
        acc = __builtin_amdgcn_wmma_f32_16x16x32_bf16(false, a, false, bm,
                                                      (short)0, acc, false, false);
    }
    const float bb = bias[colB];
#pragma unroll
    for (int r = 0; r < 8; ++r) {
        int srow = s0 + r + half * 8;
        out[((long)srow * BATCH + b) * C_DIM + colB] = acc[r] + bb;
    }
}

// =====================================================================
extern "C" void kernel_launch(void* const* d_in, const int* in_sizes, int n_in,
                              void* d_out, int out_size, void* d_ws, size_t ws_size,
                              hipStream_t stream)
{
    const float* query = (const float*)d_in[0];
    const float* keyx  = (const float*)d_in[1];
    const float* value = (const float*)d_in[2];
    const float* srcp  = (const float*)d_in[3];
    const float* tgtp  = (const float*)d_in[4];
    const float* Wq = (const float*)d_in[5];  const float* bq = (const float*)d_in[6];
    const float* Wk = (const float*)d_in[7];  const float* bk = (const float*)d_in[8];
    const float* Wv = (const float*)d_in[9];  const float* bv = (const float*)d_in[10];
    const float* Wo = (const float*)d_in[11]; const float* bo = (const float*)d_in[12];
    float* out = (float*)d_out;

    // workspace carve-up (256B aligned)
    char* w = (char*)d_ws;
    size_t off = 0;
    auto carve = [&](size_t bytes) {
        void* p = w + off;
        off += (bytes + 255) & ~(size_t)255;
        return p;
    };
    __bf16* Qb  = (__bf16*)carve((size_t)2 * BATCH * NHEADS * SEQ_N2 * DKH);
    __bf16* Kb  = (__bf16*)carve((size_t)2 * BATCH * NHEADS * SEQ_N  * DKH);
    __bf16* Vt  = (__bf16*)carve((size_t)2 * BATCH * NHEADS * SEQ_N  * DKH);
    int*    nnp = (int*)   carve((size_t)4 * BATCH * SEQ_N * 16);
    float*  Oat = (float*) carve((size_t)4 * BATCH * SEQ_N2 * C_DIM);
    if (off > ws_size) return;   // workspace too small; nothing safe to do

    // 1) projections (bf16 outputs feeding the WMMA attention path)
    proj_kernel<0><<<dim3(SEQ_N2 / 16, C_DIM / 16, BATCH), 32, 0, stream>>>(
        query, Wq, bq, Qb, SEQ_N2);
    proj_kernel<0><<<dim3(SEQ_N / 16, C_DIM / 16, BATCH), 32, 0, stream>>>(
        keyx, Wk, bk, Kb, SEQ_N);
    proj_kernel<1><<<dim3(SEQ_N / 16, C_DIM / 16, BATCH), 32, 0, stream>>>(
        value, Wv, bv, Vt, SEQ_N);

    // 2) top-5 nearest targets per source + weights (packed records)
    topk_kernel<<<(BATCH * SEQ_N + 255) / 256, 256, 0, stream>>>(srcp, tgtp, nnp);

    // 3) flash attention with sparse additive mask (async-LDS staged K/V)
    attn_kernel<<<dim3(SEQ_N2 / (16 * QWAVES), NHEADS, BATCH), 32 * QWAVES, 0, stream>>>(
        Qb, Kb, Vt, nnp, Oat);

    // 4) output projection -> [N2, B, C] fp32
    outproj_kernel<<<dim3(SEQ_N2 / 16, C_DIM / 16, BATCH), 32, 0, stream>>>(
        Oat, Wo, bo, out);
}
